// CosformerMultiHeadAttention_13280038879447
// MI455X (gfx1250) — matrix operand order repaired and checked by reading, compiled, not run on hardware
//
#include <hip/hip_runtime.h>
#include <hip/hip_bf16.h>
#include <math.h>

// ---------------- problem constants ----------------
#define BB 4
#define NN 4096
#define DD 512
#define D2 1024           // 2*DD
#define BN (BB*NN)        // 16384 rows

// ---------------- WMMA vector types ----------------
typedef __attribute__((ext_vector_type(16))) __bf16 v16bf;
typedef __attribute__((ext_vector_type(8)))  __bf16 v8bf;
typedef __attribute__((ext_vector_type(8)))  float  v8f;

__device__ __forceinline__ unsigned short f2bf(float f) {
    unsigned int u = __float_as_uint(f);
    u += 0x7FFFu + ((u >> 16) & 1u);           // round-to-nearest-even
    return (unsigned short)(u >> 16);
}
__device__ __forceinline__ float bf2f(unsigned short s) {
    return __uint_as_float(((unsigned int)s) << 16);
}

__device__ __forceinline__ v16bf mk_frag(v8bf lo, v8bf hi8) {
    v16bf a;
#pragma unroll
    for (int e = 0; e < 8; ++e) { a[e] = lo[e]; a[8 + e] = hi8[e]; }
    return a;
}

// ---------------- kernel 0: f32 -> bf16 convert ----------------
__global__ void k_cvt(const float* __restrict__ src, unsigned short* __restrict__ dst, int n) {
    int i = blockIdx.x * blockDim.x + threadIdx.x;
    if (i < n) dst[i] = f2bf(src[i]);
}

// ---------------- kernel 1: y = (relu)(x @ W^T), fused cosFormer feature map ----------------
// Register-blocked: each wave computes 32 rows x 64 cols (2x4 sub-tiles, 8 accumulators).
// Block = 8 waves -> 32 rows x 512 cols (full D).
// mode 0/1 : relu + dst[row*1024 + col] = y*sin(row), dst[row*1024 + 512 + col] = y*cos(row)
// mode 2   : dst[row*512 + col] = y
__global__ void k_gemm_qkv(const unsigned short* __restrict__ xb,
                           const unsigned short* __restrict__ wb,
                           unsigned short* __restrict__ dst,
                           int mode) {
    const int lane = threadIdx.x & 31;
    const int wave = threadIdx.x >> 5;
    const int hi   = lane >> 4;
    const int l15  = lane & 15;
    const int rowbase = blockIdx.x * 32;
    const int colbase = wave * 64;

    const unsigned short* ap0 = xb + (size_t)(rowbase + l15) * DD;
    const unsigned short* ap1 = xb + (size_t)(rowbase + 16 + l15) * DD;
    const unsigned short* bp[4];
#pragma unroll
    for (int c = 0; c < 4; ++c) bp[c] = wb + (size_t)(colbase + c * 16 + l15) * DD;

    v8f acc[2][4] = {};
    for (int k0 = 0; k0 < DD; k0 += 32) {
        v16bf a0 = mk_frag(*(const v8bf*)(ap0 + k0 + hi * 8),
                           *(const v8bf*)(ap0 + k0 + 16 + hi * 8));
        v16bf a1 = mk_frag(*(const v8bf*)(ap1 + k0 + hi * 8),
                           *(const v8bf*)(ap1 + k0 + 16 + hi * 8));
#pragma unroll
        for (int c = 0; c < 4; ++c) {
            v16bf bm = mk_frag(*(const v8bf*)(bp[c] + k0 + hi * 8), *(const v8bf*)(bp[c] + k0 + 16 + hi * 8));
            acc[0][c] = __builtin_amdgcn_wmma_f32_16x16x32_bf16(false, a0, false, bm,
                                                                (short)0, acc[0][c], false, false);
            acc[1][c] = __builtin_amdgcn_wmma_f32_16x16x32_bf16(false, a1, false, bm,
                                                                (short)0, acc[1][c], false, false);
        }
    }

#pragma unroll
    for (int rt = 0; rt < 2; ++rt) {
#pragma unroll
        for (int c = 0; c < 4; ++c) {
            int col = colbase + c * 16 + l15;
            if (mode < 2) {
#pragma unroll
                for (int i = 0; i < 8; ++i) {
                    int r = rowbase + rt * 16 + hi * 8 + i;
                    int n = r & (NN - 1);
                    float y = fmaxf(acc[rt][c][i], 0.0f);
                    float idx = 1.57079632679f * (float)(n + 1) * (1.0f / (float)NN);
                    float s, co;
                    __sincosf(idx, &s, &co);
                    dst[(size_t)r * D2 + col]      = f2bf(y * s);
                    dst[(size_t)r * D2 + DD + col] = f2bf(y * co);
                }
            } else {
#pragma unroll
                for (int i = 0; i < 8; ++i) {
                    int r = rowbase + rt * 16 + hi * 8 + i;
                    dst[(size_t)r * DD + col] = f2bf(acc[rt][c][i]);
                }
            }
        }
    }
}

// ---------------- kernel 2: ksum[b][d2] = sum_n k_[b,n,d2] ----------------
__global__ void k_ksum(const unsigned short* __restrict__ ks, float* __restrict__ ksum) {
    int t = blockIdx.x * blockDim.x + threadIdx.x;   // 0..4095
    int b = t >> 10, d2 = t & (D2 - 1);
    const unsigned short* p = ks + (size_t)b * NN * D2 + d2;
    float s = 0.f;
    for (int n = 0; n < NN; ++n) s += bf2f(p[(size_t)n * D2]);
    ksum[t] = s;
}

// ---------------- kernel 3: z[row] = 1 / max(q_[row] . ksum[b], 1e-6) ----------------
__global__ void k_z(const unsigned short* __restrict__ qs, const float* __restrict__ ksum,
                    float* __restrict__ z) {
    int wave = threadIdx.x >> 5, lane = threadIdx.x & 31;
    int row = blockIdx.x * 8 + wave;                 // 0..16383
    int b = row >> 12;
    const unsigned short* q = qs + (size_t)row * D2;
    const float* kk = ksum + b * D2;
    float s = 0.f;
    for (int j = lane; j < D2; j += 32) s += bf2f(q[j]) * kk[j];
#pragma unroll
    for (int off = 16; off > 0; off >>= 1) s += __shfl_xor(s, off, 32);
    if (lane == 0) z[row] = 1.0f / fmaxf(s, 1e-6f);
}

// ---------------- kernel 4: kvT[b][m][d2] = sum_n k_[b,n,d2] * v[b,n,m] ----------------
// K dimension (n) is the row dimension in memory -> stage tiles through LDS transposed.
// Block tile: 64 d2 x 128 m; each of 8 waves computes 32x32 (2x2 sub-tiles).
__global__ void k_kv(const unsigned short* __restrict__ ks,
                     const unsigned short* __restrict__ vb,
                     unsigned short* __restrict__ kvT) {
    __shared__ __align__(16) unsigned short ksT[64 * 40];    // [d2][n], stride 40
    __shared__ __align__(16) unsigned short vtT[128 * 40];   // [m][n],  stride 40

    const int lane = threadIdx.x & 31;
    const int wave = threadIdx.x >> 5;
    const int hi = lane >> 4, l15 = lane & 15;
    const int b = blockIdx.z;
    const int d2base = blockIdx.x * 64;
    const int mbase  = blockIdx.y * 128;
    const int wd2 = (wave >> 2) * 32;     // 0 or 32
    const int wm  = (wave & 3) * 32;      // 0,32,64,96

    const int t  = threadIdx.x;
    const int kn = t >> 3;                // n within tile (both loaders)
    const int kd = (t & 7) * 8;           // 8 d2 elems  (ks tile: 32n x 64d2)
    const int vm = (t & 7) * 16;          // 16 m elems  (v tile:  32n x 128m)

    const size_t ksbase = (size_t)b * NN * D2 + d2base;
    const size_t vbbase = (size_t)b * NN * DD + mbase;

    v8f acc[2][2] = {};
    for (int n0 = 0; n0 < NN; n0 += 32) {
        {   // k_ tile -> LDS transposed [d2][n]
            const unsigned short* p = ks + ksbase + (size_t)(n0 + kn) * D2 + kd;
            union { uint4 u; unsigned short s[8]; } w;
            w.u = *(const uint4*)p;
#pragma unroll
            for (int e = 0; e < 8; ++e) ksT[(kd + e) * 40 + kn] = w.s[e];
        }
        {   // v tile -> LDS transposed [m][n]
            const unsigned short* p = vb + vbbase + (size_t)(n0 + kn) * DD + vm;
            union { uint4 u; unsigned short s[8]; } w0, w1;
            w0.u = *(const uint4*)p;
            w1.u = *(const uint4*)(p + 8);
#pragma unroll
            for (int e = 0; e < 8; ++e) {
                vtT[(vm + e) * 40 + kn]     = w0.s[e];
                vtT[(vm + 8 + e) * 40 + kn] = w1.s[e];
            }
        }
        __syncthreads();
        v16bf a[2], bm[2];
#pragma unroll
        for (int rt = 0; rt < 2; ++rt) {
            const unsigned short* ap = &ksT[(wd2 + rt * 16 + l15) * 40];
            a[rt] = mk_frag(*(const v8bf*)(ap + hi * 8), *(const v8bf*)(ap + 16 + hi * 8));
        }
#pragma unroll
        for (int c = 0; c < 2; ++c)
            bm[c] = mk_frag(*(const v8bf*)(&vtT[(wm + c * 16 + l15) * 40 + hi * 8]), *(const v8bf*)(&vtT[(wm + c * 16 + l15) * 40 + 16 + hi * 8]));
#pragma unroll
        for (int rt = 0; rt < 2; ++rt)
#pragma unroll
            for (int c = 0; c < 2; ++c)
                acc[rt][c] = __builtin_amdgcn_wmma_f32_16x16x32_bf16(false, a[rt], false, bm[c],
                                                                     (short)0, acc[rt][c], false, false);
        __syncthreads();
    }

    // store transposed: kvT[b][m][d2]; d2 contiguous across acc elements -> one b128 per sub-tile
#pragma unroll
    for (int rt = 0; rt < 2; ++rt) {
#pragma unroll
        for (int c = 0; c < 2; ++c) {
            int mg  = mbase + wm + c * 16 + l15;
            int d2g = d2base + wd2 + rt * 16 + hi * 8;
            unsigned short* outp = kvT + ((size_t)b * DD + mg) * D2 + d2g;
            union { uint4 u; unsigned short s[8]; } w;
#pragma unroll
            for (int i = 0; i < 8; ++i) w.s[i] = f2bf(acc[rt][c][i]);
            *(uint4*)outp = w.u;
        }
    }
}

// ---------------- kernel 5: out = (q_ @ kv) * z, with head-scramble permutation ----------------
// Register-blocked like k_gemm_qkv: wave = 32 rows x 64 cols, block = 32 rows x 512 cols.
__global__ void k_out(const unsigned short* __restrict__ qs,
                      const unsigned short* __restrict__ kvT,
                      const float* __restrict__ z,
                      float* __restrict__ out) {
    const int lane = threadIdx.x & 31;
    const int wave = threadIdx.x >> 5;
    const int hi = lane >> 4, l15 = lane & 15;
    const int rowbase = blockIdx.x * 32;
    const int b = rowbase >> 12;
    const int colbase = wave * 64;

    const unsigned short* ap0 = qs + (size_t)(rowbase + l15) * D2;
    const unsigned short* ap1 = qs + (size_t)(rowbase + 16 + l15) * D2;
    const unsigned short* bp[4];
#pragma unroll
    for (int c = 0; c < 4; ++c)
        bp[c] = kvT + ((size_t)b * DD + colbase + c * 16 + l15) * D2;

    v8f acc[2][4] = {};
    for (int k0 = 0; k0 < D2; k0 += 32) {
        v16bf a0 = mk_frag(*(const v8bf*)(ap0 + k0 + hi * 8),
                           *(const v8bf*)(ap0 + k0 + 16 + hi * 8));
        v16bf a1 = mk_frag(*(const v8bf*)(ap1 + k0 + hi * 8),
                           *(const v8bf*)(ap1 + k0 + 16 + hi * 8));
#pragma unroll
        for (int c = 0; c < 4; ++c) {
            v16bf bm = mk_frag(*(const v8bf*)(bp[c] + k0 + hi * 8), *(const v8bf*)(bp[c] + k0 + 16 + hi * 8));
            acc[0][c] = __builtin_amdgcn_wmma_f32_16x16x32_bf16(false, a0, false, bm,
                                                                (short)0, acc[0][c], false, false);
            acc[1][c] = __builtin_amdgcn_wmma_f32_16x16x32_bf16(false, a1, false, bm,
                                                                (short)0, acc[1][c], false, false);
        }
    }

#pragma unroll
    for (int rt = 0; rt < 2; ++rt) {
#pragma unroll
        for (int c = 0; c < 4; ++c) {
            int col = colbase + c * 16 + l15;   // m_o
#pragma unroll
            for (int i = 0; i < 8; ++i) {
                int r = rowbase + rt * 16 + hi * 8 + i;
                int n_o = r & (NN - 1);
                float val = acc[rt][c][i] * z[r];
                // inverse of: final[b,nf,jf] = out[b, (jf>>6)*512 + (nf>>3), (nf&7)*64 + (jf&63)]
                int nf = (n_o & 511) * 8 + (col >> 6);
                int jf = (n_o >> 9) * 64 + (col & 63);
                out[((size_t)b * NN + nf) * DD + jf] = val;
            }
        }
    }
}

// ---------------- host launcher ----------------
extern "C" void kernel_launch(void* const* d_in, const int* in_sizes, int n_in,
                              void* d_out, int out_size, void* d_ws, size_t ws_size,
                              hipStream_t stream) {
    const float* x  = (const float*)d_in[0];
    const float* Wq = (const float*)d_in[1];
    const float* Wk = (const float*)d_in[2];
    const float* Wv = (const float*)d_in[3];
    float* out = (float*)d_out;

    char* ws = (char*)d_ws;
    size_t off = 0;
    auto carve = [&](size_t bytes) -> void* {
        void* p = ws + off;
        off += (bytes + 255) & ~(size_t)255;
        return p;
    };
    unsigned short* xb   = (unsigned short*)carve((size_t)BN * DD * 2);      // 16 MB
    unsigned short* wqb  = (unsigned short*)carve((size_t)DD * DD * 2);
    unsigned short* wkb  = (unsigned short*)carve((size_t)DD * DD * 2);
    unsigned short* wvb  = (unsigned short*)carve((size_t)DD * DD * 2);
    unsigned short* qs   = (unsigned short*)carve((size_t)BN * D2 * 2);      // 32 MB
    unsigned short* ksb  = (unsigned short*)carve((size_t)BN * D2 * 2);      // 32 MB
    unsigned short* vbuf = (unsigned short*)carve((size_t)BN * DD * 2);      // 16 MB
    unsigned short* kvT  = (unsigned short*)carve((size_t)BB * DD * D2 * 2); // 4 MB
    float*          ksum = (float*)carve((size_t)BB * D2 * 4);
    float*          zbuf = (float*)carve((size_t)BN * 4);
    (void)ws_size; (void)n_in; (void)in_sizes; (void)out_size;

    // 0) convert inputs to bf16
    k_cvt<<<dim3((BN * DD + 255) / 256), dim3(256), 0, stream>>>(x, xb, BN * DD);
    k_cvt<<<dim3((DD * DD + 255) / 256), dim3(256), 0, stream>>>(Wq, wqb, DD * DD);
    k_cvt<<<dim3((DD * DD + 255) / 256), dim3(256), 0, stream>>>(Wk, wkb, DD * DD);
    k_cvt<<<dim3((DD * DD + 255) / 256), dim3(256), 0, stream>>>(Wv, wvb, DD * DD);

    // 1) q_, k_, v projections (WMMA, register-blocked, fused relu + sin/cos feature map)
    dim3 gGemm(BN / 32);
    k_gemm_qkv<<<gGemm, dim3(256), 0, stream>>>(xb, wqb, qs,   0);
    k_gemm_qkv<<<gGemm, dim3(256), 0, stream>>>(xb, wkb, ksb,  1);
    k_gemm_qkv<<<gGemm, dim3(256), 0, stream>>>(xb, wvb, vbuf, 2);

    // 2) column sums of k_
    k_ksum<<<dim3((BB * D2) / 256), dim3(256), 0, stream>>>(ksb, ksum);

    // 3) normalizer z
    k_z<<<dim3(BN / 8), dim3(256), 0, stream>>>(qs, ksum, zbuf);

    // 4) kv = k_^T @ v  (WMMA over K=N with LDS transpose staging), stored transposed
    k_kv<<<dim3(D2 / 64, DD / 128, BB), dim3(256), 0, stream>>>(ksb, vbuf, kvT);

    // 5) out = (q_ @ kv) * z with head-scramble permutation (WMMA, register-blocked)
    k_out<<<gGemm, dim3(256), 0, stream>>>(qs, kvT, zbuf, out);
}
